// PaTHAttentionWfreq_71562745086195
// MI455X (gfx1250) — compile-verified, matrix operand order checked by reading, not run on hardware
//
#include <hip/hip_runtime.h>
#include <hip/hip_bf16.h>
#include <math.h>

typedef _Float16 half_t;
typedef __attribute__((ext_vector_type(16))) _Float16 v16h;
typedef __attribute__((ext_vector_type(8)))  _Float16 v8h;
typedef __attribute__((ext_vector_type(8)))  float    v8f;
typedef __attribute__((ext_vector_type(4)))  unsigned int v4u;
typedef __attribute__((ext_vector_type(8)))  int v8i;
typedef __attribute__((ext_vector_type(4)))  int v4i;

#define T_SEQ 1024
#define HIDW  1024
#define NHEAD 16
#define DH    128
#define NHD   2048   // NHEAD*DH
#define OUTW  2048

// ---------------------------------------------------------------- WMMA core

__device__ __forceinline__ v8f wmma16(v16h a, v16h b, v8f c) {
  return __builtin_amdgcn_wmma_f32_16x16x32_f16(false, a, false, b, (short)0, c,
                                                false, false);
}

// 16x32 fragment from row-major storage (element (r,k) at s[r*ld+k]).
__device__ __forceinline__ v16h frag_rowmajor(const half_t* s, int ld) {
  const int lane = threadIdx.x & 31;
  const int r  = lane & 15;
  const int kb = (lane >> 4) * 8;
  v16h f;
#pragma unroll
  for (int i = 0; i < 4; ++i) {
    f[2 * i]     = s[r * ld + kb + 2 * i];
    f[2 * i + 1] = s[r * ld + kb + 2 * i + 1];
    f[8 + 2 * i]     = s[r * ld + 16 + kb + 2 * i];
    f[8 + 2 * i + 1] = s[r * ld + 16 + kb + 2 * i + 1];
  }
  return f;
}

// ---------------------------------------------------------------- TDM (Tensor Data Mover)
// 2-D tile load of 2-byte elements: global (strided rows) -> LDS, one
// tensor_load_to_lds per call.  D# built per CDNA5 ISA:
//  group0: [1:0]=count=1, [63:32]=lds_addr, [120:64]=global_addr, [127:126]=type=2
//  group1: [17:16]=data_size(2B), [79:48]=tensor_dim0, [111:80]=tensor_dim1,
//          [127:112]=tile_dim0, [143:128]=tile_dim1, [207:160]=dim0_stride
__device__ __forceinline__ void tdm_load_2d(void* lds_dst, const void* gsrc,
                                            unsigned tensor_w, unsigned tensor_h,
                                            unsigned stride_elems,
                                            unsigned tile_w, unsigned tile_h) {
  const unsigned lds_addr = (unsigned)(unsigned long long)(size_t)lds_dst;
  const unsigned long long ga = (unsigned long long)(size_t)gsrc;
  v4u g0;
  g0[0] = 1u;                                             // count=1, user mode
  g0[1] = lds_addr;                                       // lds_addr
  g0[2] = (unsigned)(ga & 0xffffffffu);                   // global_addr[31:0]
  g0[3] = (unsigned)((ga >> 32) & 0x01ffffffu) | (2u << 30);  // addr[56:32], type=2
  v8i g1;
  g1[0] = (int)(1u << 16);                                // data_size = 2 bytes
  g1[1] = (int)((tensor_w & 0xffffu) << 16);              // tensor_dim0[15:0]
  g1[2] = (int)((tensor_w >> 16) | ((tensor_h & 0xffffu) << 16));
  g1[3] = (int)((tensor_h >> 16) | (tile_w << 16));       // tile_dim0
  g1[4] = (int)tile_h;                                    // tile_dim1 (tile_dim2=0)
  g1[5] = (int)stride_elems;                              // tensor_dim0_stride[31:0]
  g1[6] = 0;
  g1[7] = 0;
  v4i z4 = {};
#if defined(__clang_major__) && (__clang_major__ >= 23)
  v8i z8 = {};
  __builtin_amdgcn_tensor_load_to_lds(g0, g1, z4, z4, z8, 0);
#else
  __builtin_amdgcn_tensor_load_to_lds(g0, g1, z4, z4, 0);
#endif
}

__device__ __forceinline__ void tdm_wait() {
  __builtin_amdgcn_s_wait_tensorcnt(0);
}

// ---------------------------------------------------------------- utilities

__global__ void cvt_f16(const float* __restrict__ in, half_t* __restrict__ out,
                        int n) {
  int i = blockIdx.x * 256 + threadIdx.x;
  if (i < n) out[i] = (half_t)in[i];
}

// 32x32 LDS-tiled transpose: out[c][r] (f16) = in[r][c] (f32)
__global__ __launch_bounds__(256)
void transpose_to_f16(const float* __restrict__ in, half_t* __restrict__ out,
                      int rows, int cols) {
  __shared__ half_t tile[32][33];
  const int bx = blockIdx.x * 32;  // col base
  const int by = blockIdx.y * 32;  // row base
  const int tx = threadIdx.x & 31, ty = threadIdx.x >> 5;
  for (int r = ty; r < 32; r += 8)
    tile[r][tx] = (half_t)in[(size_t)(by + r) * cols + bx + tx];
  __syncthreads();
  for (int r = ty; r < 32; r += 8)
    out[(size_t)(bx + r) * rows + by + tx] = tile[tx][r];
}

// In-place l2 normalization over `width` per row (block per row).
__global__ __launch_bounds__(256) void l2norm_rows(float* __restrict__ x,
                                                   int width) {
  __shared__ float red[256];
  const int row = blockIdx.x;
  float s = 0.f;
  for (int c = threadIdx.x; c < width; c += 256) {
    float v = x[(size_t)row * width + c];
    s += v * v;
  }
  red[threadIdx.x] = s;
  __syncthreads();
  for (int off = 128; off > 0; off >>= 1) {
    if (threadIdx.x < off) red[threadIdx.x] += red[threadIdx.x + off];
    __syncthreads();
  }
  const float inv = rsqrtf(red[0]);
  for (int c = threadIdx.x; c < width; c += 256)
    x[(size_t)row * width + c] *= inv;
}

// Causal depthwise conv(K=3) + SiLU on both branches, harmonic combine -> W f16
__global__ __launch_bounds__(256)
void conv_harmonic(const float* __restrict__ wAn, const float* __restrict__ wBn,
                   const float* __restrict__ convA, const float* __restrict__ convB,
                   const float* __restrict__ phi, const float* __restrict__ omega_raw,
                   half_t* __restrict__ wh) {
  int idx = blockIdx.x * 256 + threadIdx.x;
  if (idx >= T_SEQ * OUTW) return;
  int t = idx >> 11;         // /2048
  int c = idx & (OUTW - 1);
  float yA = 0.f, yB = 0.f;
#pragma unroll
  for (int i = 0; i < 3; ++i) {
    int tt = t - 2 + i;
    if (tt >= 0) {
      yA += wAn[(size_t)tt * OUTW + c] * convA[c * 3 + i];
      yB += wBn[(size_t)tt * OUTW + c] * convB[c * 3 + i];
    }
  }
  yA = yA / (1.f + __expf(-yA));
  yB = yB / (1.f + __expf(-yB));
  int h = c >> 8, r = (c >> 7) & 1, d = c & 127;
  const float OMEGA_SCALE = 0.0007669903939428206f;  // 2*pi/8192
  float om = OMEGA_SCALE * log1pf(__expf(omega_raw[r]));
  float th = phi[r] + om * (float)t;
  float val = __cosf(th) * yA + __sinf(th) * yB;
  wh[(size_t)t * NHD + (h * 2 + r) * DH + d] = (half_t)val;
}

__global__ void beta_kernel(const float* __restrict__ logits,
                            const float* __restrict__ bias,
                            float* __restrict__ beta, int n) {
  int i = blockIdx.x * 256 + threadIdx.x;
  if (i < n) {
    float v = logits[i] + bias[i & 15];
    beta[i] = 2.f / (1.f + __expf(-v));
  }
}

// ---------------------------------------------------------------- GEMM
// C[M,N] f32 = A[M,Kd] f16 (row-major, lda) x B[N,Kd]^T f16 (row-major, ldb)
__global__ __launch_bounds__(256)
void gemm_rt(const half_t* __restrict__ A, int lda,
             const half_t* __restrict__ B, int ldb,
             float* __restrict__ C, int ldc, int M, int N, int Kd) {
  __shared__ half_t As[128][32];
  __shared__ half_t Bs[128][32];
  const int tid = threadIdx.x, lane = tid & 31, wave = tid >> 5;
  const int wm = wave >> 1, wn = wave & 1;
  const int bm = blockIdx.y * 128, bn = blockIdx.x * 128;
  v8f zf = {};
  v8f acc[2][4];
#pragma unroll
  for (int i = 0; i < 2; ++i)
#pragma unroll
    for (int j = 0; j < 4; ++j) acc[i][j] = zf;

  const int loadRow = tid >> 1;
  const int loadCol = (tid & 1) * 16;
  for (int k0 = 0; k0 < Kd; k0 += 32) {
    {
      const half_t* src = &A[(size_t)(bm + loadRow) * lda + k0 + loadCol];
      *(v8h*)&As[loadRow][loadCol]     = *(const v8h*)(src);
      *(v8h*)&As[loadRow][loadCol + 8] = *(const v8h*)(src + 8);
    }
    {
      const int gr = bn + loadRow;
      if (gr < N) {
        const half_t* src = &B[(size_t)gr * ldb + k0 + loadCol];
        *(v8h*)&Bs[loadRow][loadCol]     = *(const v8h*)(src);
        *(v8h*)&Bs[loadRow][loadCol + 8] = *(const v8h*)(src + 8);
      } else {
        v8h z = {};
        *(v8h*)&Bs[loadRow][loadCol]     = z;
        *(v8h*)&Bs[loadRow][loadCol + 8] = z;
      }
    }
    if (k0 + 32 < Kd) {  // global_prefetch_b8 of next K slab
      __builtin_prefetch(&A[(size_t)(bm + loadRow) * lda + k0 + 32 + loadCol], 0, 0);
      int pr = bn + loadRow; if (pr >= N) pr = N - 1;
      __builtin_prefetch(&B[(size_t)pr * ldb + k0 + 32 + loadCol], 0, 0);
    }
    __syncthreads();
    v16h af[2], bf[4];
#pragma unroll
    for (int i = 0; i < 2; ++i) af[i] = frag_rowmajor(&As[wm * 32 + i * 16][0], 32);
#pragma unroll
    for (int j = 0; j < 4; ++j) bf[j] = frag_rowmajor(&Bs[wn * 64 + j * 16][0], 32);
#pragma unroll
    for (int i = 0; i < 2; ++i)
#pragma unroll
      for (int j = 0; j < 4; ++j) acc[i][j] = wmma16(af[i], bf[j], acc[i][j]);
    __syncthreads();
  }
  const int cc = lane & 15, rr = (lane >> 4) * 8;
#pragma unroll
  for (int i = 0; i < 2; ++i)
#pragma unroll
    for (int j = 0; j < 4; ++j)
#pragma unroll
      for (int e = 0; e < 8; ++e) {
        int row = bm + wm * 32 + i * 16 + rr + e;
        int col = bn + wn * 64 + j * 16 + cc;
        if (row < M && col < N) C[(size_t)row * ldc + col] = acc[i][j][e];
      }
}

// ---------------------------------------------------------------- QW = -(q.w) masked, f16
__global__ __launch_bounds__(256)
void qw_gemm(const half_t* __restrict__ q16, const half_t* __restrict__ wh,
             half_t* __restrict__ QW) {
  if (blockIdx.x > blockIdx.y) return;  // strictly-upper 128-blocks never read
  const int n = blockIdx.z, hq = n >> 1;
  const int bm = blockIdx.y * 128, bn = blockIdx.x * 128;
  __shared__ half_t As[128][32];
  __shared__ half_t Bs[128][32];
  const int tid = threadIdx.x, lane = tid & 31, wave = tid >> 5;
  const int wm = wave >> 1, wn = wave & 1;
  v8f zf = {};
  v8f acc[2][4];
#pragma unroll
  for (int i = 0; i < 2; ++i)
#pragma unroll
    for (int j = 0; j < 4; ++j) acc[i][j] = zf;
  const int loadRow = tid >> 1, loadCol = (tid & 1) * 16;
  for (int k0 = 0; k0 < DH; k0 += 32) {
    {
      const half_t* src = &q16[(size_t)(bm + loadRow) * HIDW + hq * DH + k0 + loadCol];
      *(v8h*)&As[loadRow][loadCol]     = *(const v8h*)(src);
      *(v8h*)&As[loadRow][loadCol + 8] = *(const v8h*)(src + 8);
    }
    {
      const half_t* src = &wh[(size_t)(bn + loadRow) * NHD + n * DH + k0 + loadCol];
      *(v8h*)&Bs[loadRow][loadCol]     = *(const v8h*)(src);
      *(v8h*)&Bs[loadRow][loadCol + 8] = *(const v8h*)(src + 8);
    }
    __syncthreads();
    v16h af[2], bf[4];
#pragma unroll
    for (int i = 0; i < 2; ++i) af[i] = frag_rowmajor(&As[wm * 32 + i * 16][0], 32);
#pragma unroll
    for (int j = 0; j < 4; ++j) bf[j] = frag_rowmajor(&Bs[wn * 64 + j * 16][0], 32);
#pragma unroll
    for (int i = 0; i < 2; ++i)
#pragma unroll
      for (int j = 0; j < 4; ++j) acc[i][j] = wmma16(af[i], bf[j], acc[i][j]);
    __syncthreads();
  }
  half_t* dst = QW + (size_t)n * T_SEQ * T_SEQ;
  const int cc = lane & 15, rr = (lane >> 4) * 8;
#pragma unroll
  for (int i = 0; i < 2; ++i)
#pragma unroll
    for (int j = 0; j < 4; ++j)
#pragma unroll
      for (int e = 0; e < 8; ++e) {
        int t = bm + wm * 32 + i * 16 + rr + e;
        int ci = bn + wn * 64 + j * 16 + cc;
        float v = (ci <= t) ? -acc[i][j][e] : 0.f;  // negated: fused subtract later
        dst[(size_t)t * T_SEQ + ci] = (half_t)v;
      }
}

// ---------------------------------------------------------------- PaTH triangular solve
// Per head: Y = (I + beta*G)^-1 (beta*strict(W K^T)), block-64 forward
// substitution.  Output stored TRANSPOSED (YT[s][i] = Y[i][s]).
// All 64x128 operand tiles staged by the Tensor Data Mover.
__global__ __launch_bounds__(256)
void path_solver(const half_t* __restrict__ wh, const half_t* __restrict__ k16,
                 const float* __restrict__ beta, half_t* __restrict__ YT) {
  const int n = blockIdx.x, hq = n >> 1;
  __shared__ half_t Wi[64][128];
  __shared__ half_t Tb[64][128];
  __shared__ half_t Lt[64][64];
  __shared__ half_t YtT[64][64];   // tile of YT (s-major)
  __shared__ half_t Rt[64][64];
  half_t* __restrict__ YTh = YT + (size_t)n * T_SEQ * T_SEQ;
  const int tid = threadIdx.x, lane = tid & 31, wave = tid >> 5;
  v8f zf = {};

  for (int ib = 0; ib < 16; ++ib) {
    const int i0 = ib * 64;
    if (wave == 0) {  // TDM: Wi tile (64 rows x 128 halves, stride NHD)
      tdm_load_2d(&Wi[0][0], &wh[(size_t)i0 * NHD + n * DH], NHD, T_SEQ, NHD, 128, 64);
      tdm_wait();
    }
    __syncthreads();

    for (int cb = 0; cb <= ib; ++cb) {
      const int s0 = cb * 64;
      if (wave == 0) {  // TDM: K tile
        tdm_load_2d(&Tb[0][0], &k16[(size_t)s0 * HIDW + hq * DH], HIDW, T_SEQ, HIDW, 128, 64);
        tdm_wait();
      }
      __syncthreads();

      // RHS block = beta_i * (Wi . K_cb^T), strict mask on the diagonal block
      v8f acc[2] = {zf, zf};
#pragma unroll
      for (int sub = 0; sub < 2; ++sub) {
        const int ti = wave + sub * 8, tm = ti >> 2, tn = ti & 3;
#pragma unroll
        for (int kk = 0; kk < 4; ++kk) {
          v16h fa = frag_rowmajor(&Wi[tm * 16][kk * 32], 128);
          v16h fb = frag_rowmajor(&Tb[tn * 16][kk * 32], 128);
          acc[sub] = wmma16(fa, fb, acc[sub]);
        }
      }
      __syncthreads();
#pragma unroll
      for (int sub = 0; sub < 2; ++sub) {
        const int ti = wave + sub * 8, tm = ti >> 2, tn = ti & 3;
#pragma unroll
        for (int e = 0; e < 8; ++e) {
          int row = tm * 16 + ((lane >> 4) * 8) + e;
          int col = tn * 16 + (lane & 15);
          float bv = beta[(size_t)(i0 + row) * NHEAD + n];
          float v = acc[sub][e] * bv;
          if (cb == ib && (s0 + col) >= (i0 + row)) v = 0.f;
          Rt[row][col] = (half_t)v;
        }
      }
      __syncthreads();

      // Off-diagonal updates: Rt -= (beta_i*Wi.Wjb^T) . Y(jb,cb)
      for (int jb = cb; jb < ib; ++jb) {
        const int j0 = jb * 64;
        if (wave == 0) {  // TDM: Wjb tile + YT tile
          tdm_load_2d(&Tb[0][0], &wh[(size_t)j0 * NHD + n * DH], NHD, T_SEQ, NHD, 128, 64);
          tdm_load_2d(&YtT[0][0], &YTh[(size_t)s0 * T_SEQ + j0], T_SEQ, T_SEQ, T_SEQ, 64, 64);
          tdm_wait();
        }
        __syncthreads();

        v8f lac[2] = {zf, zf};
#pragma unroll
        for (int sub = 0; sub < 2; ++sub) {
          const int ti = wave + sub * 8, tm = ti >> 2, tn = ti & 3;
#pragma unroll
          for (int kk = 0; kk < 4; ++kk) {
            v16h fa = frag_rowmajor(&Wi[tm * 16][kk * 32], 128);
            v16h fb = frag_rowmajor(&Tb[tn * 16][kk * 32], 128);
            lac[sub] = wmma16(fa, fb, lac[sub]);
          }
        }
#pragma unroll
        for (int sub = 0; sub < 2; ++sub) {
          const int ti = wave + sub * 8, tm = ti >> 2, tn = ti & 3;
#pragma unroll
          for (int e = 0; e < 8; ++e) {
            int row = tm * 16 + ((lane >> 4) * 8) + e;
            int col = tn * 16 + (lane & 15);
            float bv = beta[(size_t)(i0 + row) * NHEAD + n];
            Lt[row][col] = (half_t)(lac[sub][e] * bv);  // jb<ib => strict already
          }
        }
        __syncthreads();

#pragma unroll
        for (int sub = 0; sub < 2; ++sub) {
          const int ti = wave + sub * 8, tm = ti >> 2, tn = ti & 3;
          v8f u = zf;
#pragma unroll
          for (int kk = 0; kk < 2; ++kk) {
            v16h fa = frag_rowmajor(&Lt[tm * 16][kk * 32], 64);
            v16h fb = frag_rowmajor(&YtT[tn * 16][kk * 32], 64);
            u = wmma16(fa, fb, u);
          }
#pragma unroll
          for (int e = 0; e < 8; ++e) {
            int row = tm * 16 + ((lane >> 4) * 8) + e;
            int col = tn * 16 + (lane & 15);
            Rt[row][col] = (half_t)((float)Rt[row][col] - u[e]);
          }
        }
        __syncthreads();
      }

      // Diagonal block Lii = beta_i * strict(Wi.Wi^T) (recomputed, LDS-cheap)
      {
        v8f lac[2] = {zf, zf};
#pragma unroll
        for (int sub = 0; sub < 2; ++sub) {
          const int ti = wave + sub * 8, tm = ti >> 2, tn = ti & 3;
#pragma unroll
          for (int kk = 0; kk < 4; ++kk) {
            v16h fa = frag_rowmajor(&Wi[tm * 16][kk * 32], 128);
            v16h fb = frag_rowmajor(&Wi[tn * 16][kk * 32], 128);
            lac[sub] = wmma16(fa, fb, lac[sub]);
          }
        }
#pragma unroll
        for (int sub = 0; sub < 2; ++sub) {
          const int ti = wave + sub * 8, tm = ti >> 2, tn = ti & 3;
#pragma unroll
          for (int e = 0; e < 8; ++e) {
            int row = tm * 16 + ((lane >> 4) * 8) + e;
            int col = tn * 16 + (lane & 15);
            float bv = beta[(size_t)(i0 + row) * NHEAD + n];
            Lt[row][col] = (col < row) ? (half_t)(lac[sub][e] * bv) : (half_t)0.f;
          }
        }
        __syncthreads();
      }

      // Serial forward substitution: one lane per column (64 lanes = 2 waves)
      if (tid < 64) {
        const int c = tid;
        for (int a = 1; a < 64; ++a) {
          float tmp = (float)Rt[a][c];
          for (int b2 = 0; b2 < a; ++b2)
            tmp -= (float)Lt[a][b2] * (float)Rt[b2][c];
          Rt[a][c] = (half_t)tmp;
        }
      }
      __syncthreads();
      // store block transposed: YT[(s0+s)][i0+i] = Rt[i][s]
      {
        const int sr = tid >> 2;          // s index
        const int ic = (tid & 3) * 16;    // i base
        v8h a, b;
#pragma unroll
        for (int u = 0; u < 8; ++u) {
          a[u] = Rt[ic + u][sr];
          b[u] = Rt[ic + 8 + u][sr];
        }
        *(v8h*)&YTh[(size_t)(s0 + sr) * T_SEQ + i0 + ic]     = a;
        *(v8h*)&YTh[(size_t)(s0 + sr) * T_SEQ + i0 + ic + 8] = b;
      }
      __threadfence();   // later iterations re-read YT written above
      __syncthreads();
    }
  }
}

// ---------------------------------------------------------------- flash attention
// S = (Q.K^T + (-QW).Y) * D^-1/2 with causal mask, online softmax, O += P.V
// Operand tiles staged by the TDM; softmax via wave shuffles.
__global__ __launch_bounds__(256)
void attn_flash(const half_t* __restrict__ q16, const half_t* __restrict__ k16,
                const half_t* __restrict__ VT, const half_t* __restrict__ YT,
                const half_t* __restrict__ QW, float* __restrict__ O) {
  const int rb = blockIdx.x, n = blockIdx.y, hq = n >> 1;
  const int t0 = rb * 128;
  __shared__ half_t Bt[8192];      // shared staging: K(64x128)/QW(128x64)/VT(128x64)
  __shared__ half_t YTt[64][64];
  __shared__ half_t Pb[128][64];
  __shared__ float mrow[128], lrow[128], arow[128];
  const half_t* __restrict__ YTh = YT + (size_t)n * T_SEQ * T_SEQ;
  const half_t* __restrict__ QWh = QW + (size_t)n * T_SEQ * T_SEQ;
  const int tid = threadIdx.x, lane = tid & 31, wave = tid >> 5;
  const int wm = wave >> 1, wn = wave & 1;
  v8f zf = {};
  v8f oacc[2][4];
#pragma unroll
  for (int i = 0; i < 2; ++i)
#pragma unroll
    for (int j = 0; j < 4; ++j) oacc[i][j] = zf;
  if (tid < 128) { mrow[tid] = -1e30f; lrow[tid] = 0.f; }

  // Cache this wave's Q fragments (rows wave*16..+15) in VGPRs, staged via LDS.
  v16h qf[4];
#pragma unroll
  for (int hf = 0; hf < 2; ++hf) {
    __syncthreads();
    if (wave == 0) {
      tdm_load_2d(&Bt[0], &q16[(size_t)(t0 + hf * 64) * HIDW + hq * DH],
                  HIDW, T_SEQ, HIDW, 128, 64);
      tdm_wait();
    }
    __syncthreads();
    if ((wave >> 2) == hf) {
#pragma unroll
      for (int kk = 0; kk < 4; ++kk)
        qf[kk] = frag_rowmajor(&Bt[((wave & 3) * 16) * 128 + kk * 32], 128);
    }
  }
  __syncthreads();
  const float scale = 0.08838834764831845f;  // 128^-0.5
  const int cbmax = 2 * rb + 1;

  for (int cb = 0; cb <= cbmax; ++cb) {
    const int s0 = cb * 64;
    // ---- stage K tile (64 s x 128 d) via TDM
    if (wave == 0) {
      tdm_load_2d(&Bt[0], &k16[(size_t)s0 * HIDW + hq * DH], HIDW, T_SEQ, HIDW, 128, 64);
      tdm_wait();
    }
    __syncthreads();
    v8f sacc[4] = {zf, zf, zf, zf};
#pragma unroll
    for (int kk = 0; kk < 4; ++kk)
#pragma unroll
      for (int tn = 0; tn < 4; ++tn) {
        v16h fb = frag_rowmajor(&Bt[(tn * 16) * 128 + kk * 32], 128);
        sacc[tn] = wmma16(qf[kk], fb, sacc[tn]);
      }
    __syncthreads();
    // ---- the O(T^3) (-QW).Y term
    for (int jj = cb; jj <= cbmax; ++jj) {
      if (wave == 0) {  // TDM: QW tile (128 x 64) + YT tile (64 x 64)
        tdm_load_2d(&Bt[0], &QWh[(size_t)t0 * T_SEQ + jj * 64], T_SEQ, T_SEQ, T_SEQ, 64, 128);
        tdm_load_2d(&YTt[0][0], &YTh[(size_t)s0 * T_SEQ + jj * 64], T_SEQ, T_SEQ, T_SEQ, 64, 64);
        tdm_wait();
      }
      __syncthreads();
#pragma unroll
      for (int kk = 0; kk < 2; ++kk) {
        v16h fa = frag_rowmajor(&Bt[(wave * 16) * 64 + kk * 32], 64);
#pragma unroll
        for (int tn = 0; tn < 4; ++tn) {
          v16h fb = frag_rowmajor(&YTt[tn * 16][kk * 32], 64);
          sacc[tn] = wmma16(fa, fb, sacc[tn]);  // QW pre-negated
        }
      }
      __syncthreads();
    }
    // ---- online softmax: each row lives in 16 lanes of one wave
#pragma unroll
    for (int e = 0; e < 8; ++e) {
      const int row = wave * 16 + ((lane >> 4) * 8) + e;
      const int gt = t0 + row;
      float mx = -1e30f;
#pragma unroll
      for (int tn = 0; tn < 4; ++tn) {
        int gs = s0 + tn * 16 + (lane & 15);
        float vv = (gs <= gt) ? sacc[tn][e] * scale : -1e30f;
        sacc[tn][e] = vv;
        mx = fmaxf(mx, vv);
      }
#pragma unroll
      for (int m2 = 1; m2 < 16; m2 <<= 1) mx = fmaxf(mx, __shfl_xor(mx, m2, 16));
      const float oldm = mrow[row];
      const float nm = fmaxf(oldm, mx);
      float sl = 0.f;
#pragma unroll
      for (int tn = 0; tn < 4; ++tn) {
        float p = __expf(sacc[tn][e] - nm);
        Pb[row][tn * 16 + (lane & 15)] = (half_t)p;
        sl += p;
      }
#pragma unroll
      for (int m2 = 1; m2 < 16; m2 <<= 1) sl += __shfl_xor(sl, m2, 16);
      if ((lane & 15) == 0) {
        const float al = __expf(oldm - nm);
        lrow[row] = lrow[row] * al + sl;
        mrow[row] = nm;
        arow[row] = al;
      }
    }
    __syncthreads();
    // ---- rescale O accumulators while TDM stages VT tile (128 d x 64 s)
    if (wave == 0) {
      tdm_load_2d(&Bt[0], &VT[(size_t)(hq * DH) * T_SEQ + s0], T_SEQ, HIDW, T_SEQ, 64, 128);
    }
#pragma unroll
    for (int i = 0; i < 2; ++i)
#pragma unroll
      for (int j = 0; j < 4; ++j)
#pragma unroll
        for (int e = 0; e < 8; ++e) {
          int row = wm * 32 + i * 16 + ((lane >> 4) * 8) + e;
          oacc[i][j][e] *= arow[row];
        }
    if (wave == 0) tdm_wait();
    __syncthreads();
    // ---- O += P.V
#pragma unroll
    for (int kk = 0; kk < 2; ++kk)
#pragma unroll
      for (int i = 0; i < 2; ++i) {
        v16h fa = frag_rowmajor(&Pb[wm * 32 + i * 16][kk * 32], 64);
#pragma unroll
        for (int j = 0; j < 4; ++j) {
          v16h fb = frag_rowmajor(&Bt[(wn * 64 + j * 16) * 64 + kk * 32], 64);
          oacc[i][j] = wmma16(fa, fb, oacc[i][j]);
        }
      }
    __syncthreads();
  }

#pragma unroll
  for (int i = 0; i < 2; ++i)
#pragma unroll
    for (int j = 0; j < 4; ++j)
#pragma unroll
      for (int e = 0; e < 8; ++e) {
        int row = wm * 32 + i * 16 + ((lane >> 4) * 8) + e;
        int col = wn * 64 + j * 16 + (lane & 15);
        O[(size_t)(t0 + row) * NHD + n * DH + col] = oacc[i][j][e] / lrow[row];
      }
}

// ---------------------------------------------------------------- launch

extern "C" void kernel_launch(void* const* d_in, const int* in_sizes, int n_in,
                              void* d_out, int out_size, void* d_ws, size_t ws_size,
                              hipStream_t stream) {
  (void)in_sizes; (void)n_in; (void)out_size; (void)ws_size;
  const float* x     = (const float*)d_in[0];
  const float* Wq    = (const float*)d_in[1];
  const float* Wk    = (const float*)d_in[2];
  const float* Wv    = (const float*)d_in[3];
  const float* wA1   = (const float*)d_in[4];
  const float* wA2   = (const float*)d_in[5];
  const float* wB1   = (const float*)d_in[6];
  const float* wB2   = (const float*)d_in[7];
  const float* convA = (const float*)d_in[8];
  const float* convB = (const float*)d_in[9];
  const float* bt_w  = (const float*)d_in[10];
  const float* bt_b  = (const float*)d_in[11];
  const float* Wo    = (const float*)d_in[12];
  const float* phi   = (const float*)d_in[13];
  const float* omega = (const float*)d_in[14];
  float* out = (float*)d_out;

  char* ws = (char*)d_ws;
  size_t off = 0;
  auto alloc = [&](size_t bytes) -> char* {
    char* p = ws + off;
    off += (bytes + 255) & ~(size_t)255;
    return p;
  };
  const size_t TH = (size_t)T_SEQ * HIDW;
  half_t* xh    = (half_t*)alloc(TH * 2);
  half_t* Wqh   = (half_t*)alloc(TH * 2);
  half_t* Wkh   = (half_t*)alloc(TH * 2);
  half_t* Wvh   = (half_t*)alloc(TH * 2);
  half_t* wA1h  = (half_t*)alloc(32 * HIDW * 2);
  half_t* wA2h  = (half_t*)alloc((size_t)OUTW * 32 * 2);
  half_t* wB1h  = (half_t*)alloc(32 * HIDW * 2);
  half_t* wB2h  = (half_t*)alloc((size_t)OUTW * 32 * 2);
  half_t* btwh  = (half_t*)alloc(16 * HIDW * 2);
  half_t* Woh   = (half_t*)alloc((size_t)HIDW * OUTW * 2);
  float*  q     = (float*)alloc(TH * 4);
  float*  k     = (float*)alloc(TH * 4);
  float*  v     = (float*)alloc(TH * 4);
  half_t* q16   = (half_t*)alloc(TH * 2);
  half_t* k16   = (half_t*)alloc(TH * 2);
  half_t* VT    = (half_t*)alloc(TH * 2);   // v transposed, f16: [HIDW][T]
  float*  midA  = (float*)alloc((size_t)T_SEQ * 32 * 4);
  float*  midB  = (float*)alloc((size_t)T_SEQ * 32 * 4);
  half_t* midAh = (half_t*)alloc((size_t)T_SEQ * 32 * 2);
  half_t* midBh = (half_t*)alloc((size_t)T_SEQ * 32 * 2);
  float*  wAf   = (float*)alloc((size_t)T_SEQ * OUTW * 4);
  float*  wBf   = (float*)alloc((size_t)T_SEQ * OUTW * 4);
  float*  betaL = (float*)alloc((size_t)T_SEQ * 16 * 4);
  float*  beta  = (float*)alloc((size_t)T_SEQ * 16 * 4);
  half_t* wh    = (half_t*)alloc((size_t)T_SEQ * NHD * 2);
  half_t* YTb   = (half_t*)alloc((size_t)NHEAD * T_SEQ * T_SEQ * 2);
  half_t* QWb   = (half_t*)alloc((size_t)NHEAD * T_SEQ * T_SEQ * 2);
  float*  Ob    = (float*)alloc((size_t)T_SEQ * NHD * 4);
  half_t* Obh   = (half_t*)alloc((size_t)T_SEQ * NHD * 2);

  auto cvt = [&](const float* in, half_t* o, int nn) {
    cvt_f16<<<(nn + 255) / 256, 256, 0, stream>>>(in, o, nn);
  };
  cvt(x, xh, TH);
  cvt(Wq, Wqh, TH); cvt(Wk, Wkh, TH); cvt(Wv, Wvh, TH);
  cvt(wA1, wA1h, 32 * HIDW); cvt(wA2, wA2h, OUTW * 32);
  cvt(wB1, wB1h, 32 * HIDW); cvt(wB2, wB2h, OUTW * 32);
  cvt(bt_w, btwh, 16 * HIDW); cvt(Wo, Woh, HIDW * OUTW);

  // projections
  gemm_rt<<<dim3(8, 8), 256, 0, stream>>>(xh, HIDW, Wqh, HIDW, q, HIDW, T_SEQ, HIDW, HIDW);
  gemm_rt<<<dim3(8, 8), 256, 0, stream>>>(xh, HIDW, Wkh, HIDW, k, HIDW, T_SEQ, HIDW, HIDW);
  gemm_rt<<<dim3(8, 8), 256, 0, stream>>>(xh, HIDW, Wvh, HIDW, v, HIDW, T_SEQ, HIDW, HIDW);
  // low-rank W branches
  gemm_rt<<<dim3(1, 8), 256, 0, stream>>>(xh, HIDW, wA1h, HIDW, midA, 32, T_SEQ, 32, HIDW);
  gemm_rt<<<dim3(1, 8), 256, 0, stream>>>(xh, HIDW, wB1h, HIDW, midB, 32, T_SEQ, 32, HIDW);
  cvt(midA, midAh, T_SEQ * 32); cvt(midB, midBh, T_SEQ * 32);
  gemm_rt<<<dim3(16, 8), 256, 0, stream>>>(midAh, 32, wA2h, 32, wAf, OUTW, T_SEQ, OUTW, 32);
  gemm_rt<<<dim3(16, 8), 256, 0, stream>>>(midBh, 32, wB2h, 32, wBf, OUTW, T_SEQ, OUTW, 32);
  l2norm_rows<<<T_SEQ, 256, 0, stream>>>(wAf, OUTW);
  l2norm_rows<<<T_SEQ, 256, 0, stream>>>(wBf, OUTW);
  // beta
  gemm_rt<<<dim3(1, 8), 256, 0, stream>>>(xh, HIDW, btwh, HIDW, betaL, 16, T_SEQ, 16, HIDW);
  beta_kernel<<<(T_SEQ * 16 + 255) / 256, 256, 0, stream>>>(betaL, bt_b, beta, T_SEQ * 16);
  // conv + silu + harmonic combine -> W (f16)
  conv_harmonic<<<(T_SEQ * OUTW) / 256, 256, 0, stream>>>(wAf, wBf, convA, convB, phi, omega, wh);

  cvt(q, q16, TH); cvt(k, k16, TH);
  transpose_to_f16<<<dim3(HIDW / 32, T_SEQ / 32), 256, 0, stream>>>(v, VT, T_SEQ, HIDW);

  // per-head PaTH pieces
  qw_gemm<<<dim3(8, 8, NHEAD), 256, 0, stream>>>(q16, wh, QWb);
  path_solver<<<NHEAD, 256, 0, stream>>>(wh, k16, beta, YTb);
  attn_flash<<<dim3(8, NHEAD), 256, 0, stream>>>(q16, k16, VT, YTb, QWb, Ob);

  // output projection
  cvt(Ob, Obh, T_SEQ * NHD);
  gemm_rt<<<dim3(8, 8), 256, 0, stream>>>(Obh, NHD, Woh, NHD, out, HIDW, T_SEQ, HIDW, NHD);
}
